// GatedDeltaNetAdapter_85237920957015
// MI455X (gfx1250) — compile-verified, matrix operand order checked by reading, not run on hardware
//
#include <hip/hip_runtime.h>
#include <hip/hip_bf16.h>
#include <math.h>

typedef __attribute__((ext_vector_type(16))) _Float16 v16h;
typedef __attribute__((ext_vector_type(8)))  float    v8f;

__device__ __forceinline__ float sigm(float x) { return 1.0f / (1.0f + __expf(-x)); }

// K-index mapping for 16-bit A/B fragments of v_wmma_f32_16x16x32_f16 (wave32):
// element e (0..15) of v16h, lane-half h (lane>>4):
//   vgpr v = e>>1, pair p = e&1;  K = (v<4 ? 2v : 8+2v) + p + 8h
__device__ __forceinline__ int kmap(int e, int h) {
    int v = e >> 1, p = e & 1;
    int base = (v < 4) ? (2 * v) : (8 + 2 * v);
    return base + p + 8 * h;
}

// ---------------------------------------------------------------- f32 -> f16
__global__ __launch_bounds__(256) void cvt_f32_f16(const float* __restrict__ in,
                                                   _Float16* __restrict__ out, int n) {
    int i = blockIdx.x * 256 + threadIdx.x;
    if (i < n) out[i] = (_Float16)in[i];
}

// ------------------------------------------------- WMMA GEMM: C = A * W^T
// A: [M,K] f16 row-major, W: [N,K] f16 row-major, C: [M,N] f32.
// Block: 256 thr = 8 waves. Wave covers 16 rows x 64 cols (4 wmma tiles).
// Block covers 128 rows x 64 cols.
__global__ __launch_bounds__(256) void gemm_f16_wmma(const _Float16* __restrict__ A,
                                                     const _Float16* __restrict__ W,
                                                     float* __restrict__ C,
                                                     int M, int N, int K) {
    const int lane = threadIdx.x & 31;
    const int wave = threadIdx.x >> 5;
    const int h    = lane >> 4;    // lane half
    const int l16  = lane & 15;
    const int m0   = blockIdx.x * 128 + wave * 16;
    const int n0   = blockIdx.y * 64;

    v8f acc[4] = {};
    for (int k0 = 0; k0 < K; k0 += 32) {
        v16h a;
        const _Float16* arow = A + (size_t)(m0 + l16) * K + k0;
#pragma unroll
        for (int e = 0; e < 16; ++e) a[e] = arow[kmap(e, h)];
#pragma unroll
        for (int nt = 0; nt < 4; ++nt) {
            const _Float16* wrow = W + (size_t)(n0 + nt * 16 + l16) * K + k0;
            v16h b;
#pragma unroll
            for (int e = 0; e < 16; ++e) b[e] = wrow[kmap(e, h)];
            acc[nt] = __builtin_amdgcn_wmma_f32_16x16x32_f16(
                false, a, false, b, (short)0, acc[nt], false, false);
        }
    }
    // C/D layout: VGPR r -> M = m0 + 8h + r, N = n0 + nt*16 + l16
#pragma unroll
    for (int nt = 0; nt < 4; ++nt)
#pragma unroll
        for (int r = 0; r < 8; ++r)
            C[(size_t)(m0 + 8 * h + r) * N + n0 + nt * 16 + l16] = acc[nt][r];
}

// --------------------------------------- beta / g scalar projections (1 row/block)
__global__ __launch_bounds__(256) void beta_g_kernel(const float* __restrict__ x,
                                                     const float* __restrict__ bw,
                                                     const float* __restrict__ aw,
                                                     const float* __restrict__ A_log,
                                                     const float* __restrict__ dt_bias,
                                                     float* __restrict__ beta,
                                                     float* __restrict__ g, int D) {
    const int row = blockIdx.x;
    const float* xr = x + (size_t)row * D;
    __shared__ float rb[256], ra[256];
    float sb = 0.f, sa = 0.f;
    for (int d = threadIdx.x; d < D; d += 256) {
        float xv = xr[d];
        sb += xv * bw[d];
        sa += xv * aw[d];
    }
    rb[threadIdx.x] = sb; ra[threadIdx.x] = sa;
    __syncthreads();
    for (int st = 128; st > 0; st >>= 1) {
        if (threadIdx.x < st) { rb[threadIdx.x] += rb[threadIdx.x + st];
                                ra[threadIdx.x] += ra[threadIdx.x + st]; }
        __syncthreads();
    }
    if (threadIdx.x == 0) {
        float z  = ra[0] + dt_bias[0];
        float sp = (z > 20.f) ? z : log1pf(__expf(z));        // softplus
        g[row]    = -__expf(A_log[0]) * sp;
        beta[row] = sigm(rb[0]);
    }
}

// ------------------------------------- causal depthwise conv (K=4) + SiLU
__global__ __launch_bounds__(256) void conv_silu_kernel(const float* __restrict__ in,
                                                        const float* __restrict__ w,
                                                        float* __restrict__ out,
                                                        int T, int D, int total) {
    int idx = blockIdx.x * 256 + threadIdx.x;
    if (idx >= total) return;
    int d  = idx % D;
    int bt = idx / D;
    int t  = bt % T;
    int b  = bt / T;
    float acc = 0.f;
#pragma unroll
    for (int i = 0; i < 4; ++i) {
        int ts = t - 3 + i;
        if (ts >= 0) acc += in[((size_t)b * T + ts) * D + d] * w[d * 4 + i];
    }
    out[idx] = acc * sigm(acc);
}

// --------------------------------------------- row L2 norm (in-place), * scale
__global__ __launch_bounds__(256) void l2norm_kernel(float* __restrict__ data,
                                                     int D, float scale) {
    const int row = blockIdx.x;
    float* r = data + (size_t)row * D;
    __shared__ float red[256];
    float s = 0.f;
    for (int d = threadIdx.x; d < D; d += 256) { float v = r[d]; s += v * v; }
    red[threadIdx.x] = s;
    __syncthreads();
    for (int st = 128; st > 0; st >>= 1) {
        if (threadIdx.x < st) red[threadIdx.x] += red[threadIdx.x + st];
        __syncthreads();
    }
    float inv = rsqrtf(red[0] + 1e-6f) * scale;
    for (int d = threadIdx.x; d < D; d += 256) r[d] *= inv;
}

// ------------------------------------------------------- gated delta scan
// Block handles (batch b, 64-column v-slice). 256 thr = 4 row-groups x 64 cols.
// Each thread owns 128 rows of S for its column: registers.
__global__ __launch_bounds__(256) void scan_kernel(const float* __restrict__ q,
                                                   const float* __restrict__ k,
                                                   const float* __restrict__ v,
                                                   const float* __restrict__ g,
                                                   const float* __restrict__ beta,
                                                   float* __restrict__ o, int T) {
    const int D   = 512;
    const int tid = threadIdx.x;
    const int c   = tid & 63;             // column within slice
    const int r   = tid >> 6;             // row group 0..3
    const int b   = blockIdx.y;
    const int j   = blockIdx.x * 64 + c;  // global column
    const int k0  = r * 128;

    __shared__ float sh_k[512];
    __shared__ float sh_q[512];
    __shared__ float sh_red[4][64];
    __shared__ float sh_dv[64];

    float s[128];
#pragma unroll
    for (int i = 0; i < 128; ++i) s[i] = 0.f;

    const size_t base = (size_t)b * T;
    for (int t = 0; t < T; ++t) {
        const size_t row = base + t;
        __syncthreads();                          // LDS reuse fence
        const float* krow = k + row * D;
        const float* qrow = q + row * D;
        sh_k[tid]       = krow[tid];
        sh_k[tid + 256] = krow[tid + 256];
        sh_q[tid]       = qrow[tid];
        sh_q[tid + 256] = qrow[tid + 256];
        __syncthreads();

        const float decay = __expf(g[row]);

        float pkv = 0.f;
#pragma unroll
        for (int i = 0; i < 128; ++i) pkv += sh_k[k0 + i] * s[i];
        sh_red[r][c] = pkv;
        __syncthreads();
        if (r == 0) {
            float kv  = decay * (sh_red[0][c] + sh_red[1][c] + sh_red[2][c] + sh_red[3][c]);
            float v_t = v[row * D + j];
            sh_dv[c]  = (v_t - kv) * beta[row];
        }
        __syncthreads();
        const float dv = sh_dv[c];

        float po = 0.f;
#pragma unroll
        for (int i = 0; i < 128; ++i) {
            float sn = s[i] * decay + sh_k[k0 + i] * dv;  // fused decay + rank-1
            s[i] = sn;
            po += sh_q[k0 + i] * sn;
        }
        sh_red[r][c] = po;
        __syncthreads();
        if (r == 0)
            o[row * D + j] = sh_red[0][c] + sh_red[1][c] + sh_red[2][c] + sh_red[3][c];
    }
}

// --------------------------- gated RMSNorm, SiLU gate, emit f16 for out-proj
__global__ __launch_bounds__(256) void rmsnorm_gate_kernel(const float* __restrict__ o,
                                                           const float* __restrict__ gp,
                                                           const float* __restrict__ nw,
                                                           _Float16* __restrict__ outh,
                                                           int D) {
    const int row = blockIdx.x;
    const float* orow = o  + (size_t)row * D;
    const float* grow = gp + (size_t)row * D;
    __shared__ float red[256];
    float s = 0.f;
    for (int d = threadIdx.x; d < D; d += 256) { float v = orow[d]; s += v * v; }
    red[threadIdx.x] = s;
    __syncthreads();
    for (int st = 128; st > 0; st >>= 1) {
        if (threadIdx.x < st) red[threadIdx.x] += red[threadIdx.x + st];
        __syncthreads();
    }
    float inv = rsqrtf(red[0] / (float)D + 1e-5f);
    for (int d = threadIdx.x; d < D; d += 256) {
        float on = orow[d] * inv * nw[d];
        float gt = grow[d];
        outh[(size_t)row * D + d] = (_Float16)(on * gt * sigm(gt));
    }
}

// ===========================================================================
extern "C" void kernel_launch(void* const* d_in, const int* in_sizes, int n_in,
                              void* d_out, int out_size, void* d_ws, size_t ws_size,
                              hipStream_t stream) {
    const int B = 2, T = 1024, D = 512;
    const int M = B * T;                 // 2048 rows

    const float* x    = (const float*)d_in[0];
    const float* wq   = (const float*)d_in[1];
    const float* wk   = (const float*)d_in[2];
    const float* wv   = (const float*)d_in[3];
    const float* bw   = (const float*)d_in[4];
    const float* aw   = (const float*)d_in[5];
    const float* Alog = (const float*)d_in[6];
    const float* dtb  = (const float*)d_in[7];
    const float* qcw  = (const float*)d_in[8];
    const float* kcw  = (const float*)d_in[9];
    const float* vcw  = (const float*)d_in[10];
    const float* wg   = (const float*)d_in[11];
    const float* onw  = (const float*)d_in[12];
    const float* wo   = (const float*)d_in[13];
    float* out = (float*)d_out;

    char* ws = (char*)d_ws;
    size_t off = 0;
    auto alloc = [&](size_t bytes) -> void* {
        void* p = ws + off;
        off = (off + bytes + 255) & ~(size_t)255;
        return p;
    };
    _Float16* hx  = (_Float16*)alloc((size_t)M * D * 2);
    _Float16* hwq = (_Float16*)alloc((size_t)D * D * 2);
    _Float16* hwk = (_Float16*)alloc((size_t)D * D * 2);
    _Float16* hwv = (_Float16*)alloc((size_t)D * D * 2);
    _Float16* hwg = (_Float16*)alloc((size_t)D * D * 2);
    _Float16* hwo = (_Float16*)alloc((size_t)D * D * 2);
    float* qp = (float*)alloc((size_t)M * D * 4);
    float* kp = (float*)alloc((size_t)M * D * 4);
    float* vp = (float*)alloc((size_t)M * D * 4);
    float* gp = (float*)alloc((size_t)M * D * 4);
    float* qc = (float*)alloc((size_t)M * D * 4);
    float* kc = (float*)alloc((size_t)M * D * 4);
    float* vc = (float*)alloc((size_t)M * D * 4);
    float* bb = (float*)alloc((size_t)M * 4);
    float* gb = (float*)alloc((size_t)M * 4);
    float* ob = (float*)alloc((size_t)M * D * 4);
    _Float16* ho = (_Float16*)alloc((size_t)M * D * 2);

    // 1) convert inputs to f16 for WMMA GEMMs
    cvt_f32_f16<<<(M * D + 255) / 256, 256, 0, stream>>>(x, hx, M * D);
    cvt_f32_f16<<<(D * D + 255) / 256, 256, 0, stream>>>(wq, hwq, D * D);
    cvt_f32_f16<<<(D * D + 255) / 256, 256, 0, stream>>>(wk, hwk, D * D);
    cvt_f32_f16<<<(D * D + 255) / 256, 256, 0, stream>>>(wv, hwv, D * D);
    cvt_f32_f16<<<(D * D + 255) / 256, 256, 0, stream>>>(wg, hwg, D * D);
    cvt_f32_f16<<<(D * D + 255) / 256, 256, 0, stream>>>(wo, hwo, D * D);

    // 2) q/k/v/gate projections via WMMA
    dim3 gg(M / 128, D / 64);
    gemm_f16_wmma<<<gg, 256, 0, stream>>>(hx, hwq, qp, M, D, D);
    gemm_f16_wmma<<<gg, 256, 0, stream>>>(hx, hwk, kp, M, D, D);
    gemm_f16_wmma<<<gg, 256, 0, stream>>>(hx, hwv, vp, M, D, D);
    gemm_f16_wmma<<<gg, 256, 0, stream>>>(hx, hwg, gp, M, D, D);

    // 3) beta / g scalars (full f32 path)
    beta_g_kernel<<<M, 256, 0, stream>>>(x, bw, aw, Alog, dtb, bb, gb, D);

    // 4) causal depthwise conv + SiLU
    int total = M * D;
    conv_silu_kernel<<<(total + 255) / 256, 256, 0, stream>>>(qp, qcw, qc, T, D, total);
    conv_silu_kernel<<<(total + 255) / 256, 256, 0, stream>>>(kp, kcw, kc, T, D, total);
    conv_silu_kernel<<<(total + 255) / 256, 256, 0, stream>>>(vp, vcw, vc, T, D, total);

    // 5) L2 norm rows; q also scaled by Dh^-0.5
    l2norm_kernel<<<M, 256, 0, stream>>>(qc, D, 0.044194173824159216f); // 512^-0.5
    l2norm_kernel<<<M, 256, 0, stream>>>(kc, D, 1.0f);

    // 6) sequential gated-delta scan; state register-resident per block
    scan_kernel<<<dim3(D / 64, B), 256, 0, stream>>>(qc, kc, vc, gb, bb, ob, T);

    // 7) gated RMSNorm -> f16
    rmsnorm_gate_kernel<<<M, 256, 0, stream>>>(ob, gp, onw, ho, D);

    // 8) output projection via WMMA -> f32 out
    gemm_f16_wmma<<<gg, 256, 0, stream>>>(ho, hwo, out, M, D, D);
}
